// Backbone_78958678769789
// MI455X (gfx1250) — matrix operand order repaired and checked by reading, not compiled, not run on hardware
//
#include <hip/hip_runtime.h>
#include <stdint.h>

#define CCH 256
#define NPT 32768
#define BB  2
#define HH  128
#define WWD 128
#define HWC 16384
#define XTS 264   // LDS k-stride (elements); 264*2B = 528B = 16B-aligned per column

typedef __attribute__((ext_vector_type(16))) __bf16 v16bf;
typedef __attribute__((ext_vector_type(8)))  float  v8f;

union Frag { v16bf v; uint4 q[2]; };

__device__ __forceinline__ unsigned short f2bf(float f) {
    union { float f; unsigned u; } x; x.f = f;
    unsigned r = x.u + 0x7FFFu + ((x.u >> 16) & 1u);
    return (unsigned short)(r >> 16);
}
__device__ __forceinline__ int clampcell(int c) {
    return c < 0 ? 0 : (c > HWC - 1 ? HWC - 1 : c);
}

// ---------------- projection weights ----------------
__global__ void k_counts(const int* __restrict__ ci, const float* __restrict__ occ,
                         float* __restrict__ counts, int g) {
    int i = blockIdx.x * 256 + threadIdx.x;     // BB*NPT
    int b = i / NPT, n = i % NPT;
    int cell = clampcell(ci[(b * 3 + g) * NPT + n]);
    float o = occ[b * NPT + n];
    atomicAdd(&counts[b * HWC + cell], o * o);
}

__global__ void k_wpt(const int* __restrict__ ci, const float* __restrict__ occ,
                      const float* __restrict__ counts, float* __restrict__ wpt, int g) {
    int i = blockIdx.x * 256 + threadIdx.x;
    int b = i / NPT, n = i % NPT;
    int cell = clampcell(ci[(b * 3 + g) * NPT + n]);
    float o = occ[b * NPT + n];
    float npp = o * counts[b * HWC + cell];
    wpt[(g * BB + b) * NPT + n] = o / (npp + 1e-6f);
}

// ---------------- SpatialMix: LN over C + weight + scatter-add ----------------
__global__ void k_ln_scatter(const float* __restrict__ tok, const int* __restrict__ ci_all,
                             const float* __restrict__ wpt,
                             const float* __restrict__ lg, const float* __restrict__ lb,
                             float* __restrict__ grid, int g) {
    __shared__ float psum[256], psq[256], cmean[64], crstd[64], cwp[64];
    __shared__ int   ccell[64];
    int b  = blockIdx.x / (NPT / 64);
    int n0 = (blockIdx.x % (NPT / 64)) * 64;
    int t = threadIdx.x, col = t & 63, q = t >> 6;
    const float* base = tok + (size_t)b * CCH * NPT + n0 + col;
    float s = 0.f, ss = 0.f;
    for (int k = 0; k < 64; ++k) {
        float x = base[(size_t)(q * 64 + k) * NPT];
        s += x; ss += x * x;
    }
    psum[t] = s; psq[t] = ss;
    __syncthreads();
    if (t < 64) {
        float S = psum[t] + psum[t + 64] + psum[t + 128] + psum[t + 192];
        float Q = psq[t]  + psq[t + 64]  + psq[t + 128]  + psq[t + 192];
        float m = S * (1.f / CCH);
        float v = Q * (1.f / CCH) - m * m;
        cmean[t] = m;
        crstd[t] = rsqrtf(v + 1e-5f);
        ccell[t] = clampcell(ci_all[(b * 3 + g) * NPT + n0 + t]);
        cwp[t]   = wpt[(g * BB + b) * NPT + n0 + t];
    }
    __syncthreads();
    float m = cmean[col], r = crstd[col], wp = cwp[col];
    float* gbase = grid + (size_t)b * CCH * HWC + ccell[col];
    for (int k = 0; k < 64; ++k) {
        int c = q * 64 + k;
        float x = base[(size_t)c * NPT];
        float v = ((x - m) * r * lg[c] + lb[c]) * wp;
        atomicAdd(gbase + (size_t)c * HWC, v);
    }
}

// ---------------- depthwise 3x3 conv, SAME ----------------
__global__ void k_dwconv3(const float* __restrict__ in, float* __restrict__ out,
                          const float* __restrict__ wt, const float* __restrict__ bs,
                          int do_relu) {
    int i = blockIdx.x * 256 + threadIdx.x;      // BB*CCH*HWC
    int x  = i & (WWD - 1);
    int y  = (i >> 7) & (HH - 1);
    int bc = i >> 14;
    int c  = bc & (CCH - 1);
    const float* w = wt + c * 9;
    const float* p = in + (size_t)bc * HWC;
    float acc = bs[c];
#pragma unroll
    for (int dy = -1; dy <= 1; ++dy) {
#pragma unroll
        for (int dx = -1; dx <= 1; ++dx) {
            int yy = y + dy, xx = x + dx;
            if (yy >= 0 && yy < HH && xx >= 0 && xx < WWD)
                acc += w[(dy + 1) * 3 + (dx + 1)] * p[yy * WWD + xx];
        }
    }
    if (do_relu) acc = fmaxf(acc, 0.f);
    out[i] = acc;
}

// ---------------- BatchNorm (train) stats per channel ----------------
__global__ void k_bnstats(const float* __restrict__ plane, const float* __restrict__ asc,
                          float* __restrict__ stats) {
    int c = blockIdx.x, t = threadIdx.x;
    __shared__ float s1[256], s2[256];
    float a = asc[c], s = 0.f, ss = 0.f;
    for (int i = t; i < BB * HWC; i += 256) {
        int b = i >> 14, hw = i & (HWC - 1);
        float v = plane[((size_t)(b * CCH + c)) * HWC + hw] * a;
        s += v; ss += v * v;
    }
    s1[t] = s; s2[t] = ss;
    __syncthreads();
    for (int o = 128; o; o >>= 1) {
        if (t < o) { s1[t] += s1[t + o]; s2[t] += s2[t + o]; }
        __syncthreads();
    }
    if (t == 0) {
        float m = s1[0] * (1.f / (BB * HWC));
        float v = s2[0] * (1.f / (BB * HWC)) - m * m;
        stats[c]       = m;
        stats[CCH + c] = rsqrtf(v + 1e-5f);
    }
}

// ---------------- sigmoid gate + gather + residual ----------------
__global__ void k_gather(const float* __restrict__ plane, const int* __restrict__ ci_all,
                         const float* __restrict__ stats, const float* __restrict__ asc,
                         const float* __restrict__ bg, const float* __restrict__ bbeta,
                         float* __restrict__ tok, int g) {
    __shared__ float sm[256], sr[256], sa[256], sgm[256], sbt[256];
    int t = threadIdx.x;
    sm[t] = stats[t]; sr[t] = stats[CCH + t];
    sa[t] = asc[t];   sgm[t] = bg[t]; sbt[t] = bbeta[t];
    __syncthreads();
    int i = blockIdx.x * 256 + t;                // BB*NPT
    int b = i / NPT, n = i % NPT;
    int cell = clampcell(ci_all[(b * 3 + g) * NPT + n]);
    const float* pb = plane + (size_t)b * CCH * HWC + cell;
    float* tb = tok + (size_t)b * CCH * NPT + n;
    for (int c = 0; c < CCH; ++c) {
        float p  = pb[(size_t)c * HWC];
        float bn = (p * sa[c] - sm[c]) * sr[c] * sgm[c] + sbt[c];
        float att = 1.f / (1.f + __expf(-bn));
        tb[(size_t)c * NPT] += att * p;
    }
}

// ---------------- weight f32 -> bf16 prep ----------------
__global__ void k_wbf(const float* __restrict__ w1, const float* __restrict__ w2,
                      unsigned short* __restrict__ o1, unsigned short* __restrict__ o2) {
    int i = blockIdx.x * 256 + threadIdx.x;
    if (i < CCH * CCH) { o1[i] = f2bf(w1[i]); o2[i] = f2bf(w2[i]); }
}

// ---------------- ChannelMix GEMM1: h = relu(W1 @ LN(tok) + b1), bf16 out ----------------
__global__ void k_gemm1(const float* __restrict__ tok, const float* __restrict__ lg,
                        const float* __restrict__ lb, const unsigned short* __restrict__ wbf,
                        const float* __restrict__ b1, unsigned short* __restrict__ hbf) {
    __shared__ unsigned short xt[64 * XTS];
    __shared__ float psum[256], psq[256], cmean[64], crstd[64];
    int b  = blockIdx.x / (NPT / 64);
    int n0 = (blockIdx.x % (NPT / 64)) * 64;
    int t = threadIdx.x, col = t & 63, q = t >> 6;
    const float* base = tok + (size_t)b * CCH * NPT + n0 + col;
    float s = 0.f, ss = 0.f;
    for (int k = 0; k < 64; ++k) {
        float x = base[(size_t)(q * 64 + k) * NPT];
        s += x; ss += x * x;
    }
    psum[t] = s; psq[t] = ss;
    __syncthreads();
    if (t < 64) {
        float S = psum[t] + psum[t + 64] + psum[t + 128] + psum[t + 192];
        float Q = psq[t]  + psq[t + 64]  + psq[t + 128]  + psq[t + 192];
        float m = S * (1.f / CCH);
        float v = Q * (1.f / CCH) - m * m;
        cmean[t] = m; crstd[t] = rsqrtf(v + 1e-5f);
    }
    __syncthreads();
    {
        float m = cmean[col], r = crstd[col];
        for (int k = 0; k < 64; ++k) {
            int c = q * 64 + k;
            float x = base[(size_t)c * NPT];
            xt[col * XTS + c] = f2bf((x - m) * r * lg[c] + lb[c]);
        }
    }
    __syncthreads();
    int wave = t >> 5, lane = t & 31;
    int ct = wave & 3, rhalf = wave >> 2;
    int lm = lane & 15, lhi = lane >> 4;
    for (int i = 0; i < 8; ++i) {
        int rt = rhalf * 8 + i;
        v8f acc = {0.f, 0.f, 0.f, 0.f, 0.f, 0.f, 0.f, 0.f};
#pragma unroll
        for (int kt = 0; kt < 8; ++kt) {
            Frag fa, fb;
            const unsigned short* ap = wbf + (size_t)(rt * 16 + lm) * CCH + kt * 32 + lhi * 8;
            fa.q[0] = *(const uint4*)ap;
            fa.q[1] = *(const uint4*)(ap + 16);
            const unsigned short* bp = xt + (size_t)(ct * 16 + lm) * XTS + kt * 32 + lhi * 8;
            fb.q[0] = *(const uint4*)bp;
            fb.q[1] = *(const uint4*)(bp + 16);
            acc = __builtin_amdgcn_wmma_f32_16x16x32_bf16(
                false, fa.v, false, fb.v, (short)0, acc, false, false);
        }
        int n = n0 + ct * 16 + lm;
#pragma unroll
        for (int v = 0; v < 8; ++v) {
            int mrow = rt * 16 + v + lhi * 8;
            float h = fmaxf(acc[v] + b1[mrow], 0.f);
            hbf[((size_t)(b * CCH + mrow)) * NPT + n] = f2bf(h);
        }
    }
}

// ---------------- ChannelMix GEMM2: tok += cmix_scale * (W2 @ h + b2) ----------------
__global__ void k_gemm2(const unsigned short* __restrict__ hbf,
                        const unsigned short* __restrict__ wbf,
                        const float* __restrict__ b2, const float* __restrict__ cs,
                        float* __restrict__ tok) {
    __shared__ unsigned short xt[64 * XTS];
    int b  = blockIdx.x / (NPT / 64);
    int n0 = (blockIdx.x % (NPT / 64)) * 64;
    int t = threadIdx.x, col = t & 63, q = t >> 6;
    const unsigned short* hb = hbf + (size_t)b * CCH * NPT + n0 + col;
    for (int k = 0; k < 64; ++k) {
        int c = q * 64 + k;
        xt[col * XTS + c] = hb[(size_t)c * NPT];
    }
    __syncthreads();
    int wave = t >> 5, lane = t & 31;
    int ct = wave & 3, rhalf = wave >> 2;
    int lm = lane & 15, lhi = lane >> 4;
    for (int i = 0; i < 8; ++i) {
        int rt = rhalf * 8 + i;
        v8f acc = {0.f, 0.f, 0.f, 0.f, 0.f, 0.f, 0.f, 0.f};
#pragma unroll
        for (int kt = 0; kt < 8; ++kt) {
            Frag fa, fb;
            const unsigned short* ap = wbf + (size_t)(rt * 16 + lm) * CCH + kt * 32 + lhi * 8;
            fa.q[0] = *(const uint4*)ap;
            fa.q[1] = *(const uint4*)(ap + 16);
            const unsigned short* bp = xt + (size_t)(ct * 16 + lm) * XTS + kt * 32 + lhi * 8;
            fb.q[0] = *(const uint4*)bp;
            fb.q[1] = *(const uint4*)(bp + 16);
            acc = __builtin_amdgcn_wmma_f32_16x16x32_bf16(
                false, fa.v, false, fb.v, (short)0, acc, false, false);
        }
        int n = n0 + ct * 16 + lm;
#pragma unroll
        for (int v = 0; v < 8; ++v) {
            int mrow = rt * 16 + v + lhi * 8;
            float val = acc[v] + b2[mrow];
            tok[((size_t)(b * CCH + mrow)) * NPT + n] += cs[mrow] * val;
        }
    }
}

extern "C" void kernel_launch(void* const* d_in, const int* in_sizes, int n_in,
                              void* d_out, int out_size, void* d_ws, size_t ws_size,
                              hipStream_t stream) {
    const float* tokens_in = (const float*)d_in[0];
    const int*   cell_ind  = (const int*)d_in[1];
    const float* occ       = (const float*)d_in[2];
    const float* smix_ln_g = (const float*)d_in[3];
    const float* smix_ln_b = (const float*)d_in[4];
    const float* ffn_w1    = (const float*)d_in[5];
    const float* ffn_b1    = (const float*)d_in[6];
    const float* ffn_w2    = (const float*)d_in[7];
    const float* ffn_b2    = (const float*)d_in[8];
    const float* att_scale = (const float*)d_in[9];
    const float* att_bn_g  = (const float*)d_in[10];
    const float* att_bn_b  = (const float*)d_in[11];
    const float* cmix_ln_g = (const float*)d_in[12];
    const float* cmix_ln_b = (const float*)d_in[13];
    const float* cmix_w1   = (const float*)d_in[14];
    const float* cmix_b1   = (const float*)d_in[15];
    const float* cmix_w2   = (const float*)d_in[16];
    const float* cmix_b2   = (const float*)d_in[17];
    const float* cmix_sc   = (const float*)d_in[18];
    (void)in_sizes; (void)n_in; (void)out_size; (void)ws_size;
    float* tok = (float*)d_out;

    uint8_t* wsp = (uint8_t*)d_ws;
    auto carve = [&](size_t bytes) -> void* {
        void* p = (void*)wsp;
        wsp += (bytes + 255) & ~(size_t)255;
        return p;
    };
    const size_t planeBytes = (size_t)BB * CCH * HWC * 4;   // 32 MB
    float* grid   = (float*)carve(planeBytes);
    float* ybuf   = (float*)carve(planeBytes);
    float* planeb = (float*)carve(planeBytes);
    float* skip0  = (float*)carve(planeBytes);              // plane of layer 0 (reused at d=3)
    unsigned short* hbf = (unsigned short*)carve((size_t)BB * CCH * NPT * 2);
    float* wptb   = (float*)carve((size_t)3 * BB * NPT * 4);
    float* counts = (float*)carve((size_t)BB * HWC * 4);
    unsigned short* w1bf = (unsigned short*)carve((size_t)CCH * CCH * 2);
    unsigned short* w2bf = (unsigned short*)carve((size_t)CCH * CCH * 2);
    float* stats  = (float*)carve((size_t)2 * CCH * 4);

    // working tokens live in d_out
    hipMemcpyAsync(tok, tokens_in, (size_t)BB * CCH * NPT * 4,
                   hipMemcpyDeviceToDevice, stream);

    for (int g = 0; g < 3; ++g) {
        hipMemsetAsync(counts, 0, (size_t)BB * HWC * 4, stream);
        k_counts<<<BB * NPT / 256, 256, 0, stream>>>(cell_ind, occ, counts, g);
        k_wpt<<<BB * NPT / 256, 256, 0, stream>>>(cell_ind, occ, counts, wptb, g);
    }

    for (int d = 0; d < 4; ++d) {
        int g = d % 3;
        if (d == 3)
            hipMemcpyAsync(grid, skip0, planeBytes, hipMemcpyDeviceToDevice, stream);
        else
            hipMemsetAsync(grid, 0, planeBytes, stream);

        k_ln_scatter<<<BB * (NPT / 64), 256, 0, stream>>>(
            tok, cell_ind, wptb, smix_ln_g + d * CCH, smix_ln_b + d * CCH, grid, g);

        k_dwconv3<<<(BB * CCH * HWC) / 256, 256, 0, stream>>>(
            grid, ybuf, ffn_w1 + d * CCH * 9, ffn_b1 + d * CCH, 1);
        float* plane = (d == 0) ? skip0 : planeb;
        k_dwconv3<<<(BB * CCH * HWC) / 256, 256, 0, stream>>>(
            ybuf, plane, ffn_w2 + d * CCH * 9, ffn_b2 + d * CCH, 0);

        k_bnstats<<<CCH, 256, 0, stream>>>(plane, att_scale + d * CCH, stats);
        k_gather<<<BB * NPT / 256, 256, 0, stream>>>(
            plane, cell_ind, stats, att_scale + d * CCH,
            att_bn_g + d * CCH, att_bn_b + d * CCH, tok, g);

        k_wbf<<<(CCH * CCH) / 256, 256, 0, stream>>>(
            cmix_w1 + d * CCH * CCH, cmix_w2 + d * CCH * CCH, w1bf, w2bf);
        k_gemm1<<<BB * (NPT / 64), 256, 0, stream>>>(
            tok, cmix_ln_g + d * CCH, cmix_ln_b + d * CCH, w1bf, cmix_b1 + d * CCH, hbf);
        k_gemm2<<<BB * (NPT / 64), 256, 0, stream>>>(
            hbf, w2bf, cmix_b2 + d * CCH, cmix_sc + d * CCH, tok);
    }
}
